// EEG_CNN_50732153700698
// MI455X (gfx1250) — matrix operand order repaired and checked by reading, not compiled, not run on hardware
//
#include <hip/hip_runtime.h>

// ---------------------------------------------------------------------------
// CDNA5 / gfx1250 fused SNN (EEG CNN) implementation.
//   Kernel 1: conv1 (1x32, pad 16) + BN folded  -> c1[b, k=c*22+h, w] in d_ws
//   Kernel 2: one block per batch element; full T=64 LIF scan fused in LDS.
//             conv2 (groups=8, 22 taps) is done as a dense 16x(176->192)x64
//             f16 WMMA GEMM on binary spikes (exact in f16), f32 accumulate.
// ---------------------------------------------------------------------------

typedef __attribute__((ext_vector_type(16))) _Float16     v16h;
typedef __attribute__((ext_vector_type(8)))  float        v8f;
typedef __attribute__((ext_vector_type(4)))  unsigned int uvec4;

#define BN_EPS 1e-5f
#define NB     128      // batch
#define NK     176      // 8 channels * 22 rows (GEMM K, padded to 192)
#define NKP    192
#define NW     64       // conv1 output width
#define TSTEPS 64

// ---------------- Kernel 1: conv1 + BN ----------------
__global__ __launch_bounds__(256) void eeg_conv1_bn(
    const float* __restrict__ x,  const float* __restrict__ w1,
    const float* __restrict__ g1, const float* __restrict__ b1,
    const float* __restrict__ m1, const float* __restrict__ v1,
    float* __restrict__ c1out)
{
  int idx = blockIdx.x * 256 + threadIdx.x;
  if (idx >= NB * NK * NW) return;
  int w = idx & 63;
  int k = (idx >> 6) % NK;
  int b = idx / (NK * NW);
  int c = k / 22, h = k - c * 22;
  float scale = g1[c] * rsqrtf(v1[c] + BN_EPS);
  float shift = b1[c] - m1[c] * scale;
  const float* xrow = x + (b * 22 + h) * 63;
  float acc = 0.f;
#pragma unroll
  for (int j = 0; j < 32; ++j) {            // cross-correlation, pad 16
    int wi = w + j - 16;
    float xv = (wi >= 0 && wi < 63) ? xrow[wi] : 0.f;
    acc = fmaf(xv, w1[c * 32 + j], acc);
  }
  c1out[idx] = fmaf(acc, scale, shift);
}

// ---------------- dynamic-LDS carve-out (bytes) ----------------
#define OFF_C1    0              // 11264 f32  (45056 B)
#define OFF_V1    45056          // 11264 f32
#define OFF_S1T   90112          // 64*200 f16 (25600 B), K-major transposed spikes
#define OFF_WA    115712         // 16*192 f16 (6144 B)  dense conv2 weight (BN-scaled)
#define OFF_S2    121856         // 16*64 f32
#define OFF_P2    125952         // 16*32 f32 (8..23 live, halo zero)
#define OFF_D     128000         // 16*20 f32
#define OFF_S3    129280         // 16*20 f32
#define OFF_P3    130560         // 64 f32
#define OFF_SH2   130816         // 16 f32
#define OFF_SC3   130880         // 16 f32
#define OFF_SH3   130944         // 16 f32
#define OFF_W3D   131008         // 256 f32
#define OFF_W3P   132032         // 256 f32
#define OFF_FCW   133056         // 256 f32
#define OFF_FCB   134080         // 4 f32
#define SMEM_BYTES 134096

// ---------------- Kernel 2: fused temporal SNN ----------------
__global__ __launch_bounds__(256) void eeg_snn_fused(
    const float* __restrict__ c1g,
    const float* __restrict__ w2,  const float* __restrict__ g2,
    const float* __restrict__ b2,  const float* __restrict__ m2,
    const float* __restrict__ v2g,
    const float* __restrict__ w3d, const float* __restrict__ w3p,
    const float* __restrict__ g3,  const float* __restrict__ b3,
    const float* __restrict__ m3,  const float* __restrict__ v3g,
    const float* __restrict__ fcw, const float* __restrict__ fcb,
    float* __restrict__ out)
{
  extern __shared__ char smem[];
  float*    c1L  = (float*)(smem + OFF_C1);
  float*    v1L  = (float*)(smem + OFF_V1);
  _Float16* s1T  = (_Float16*)(smem + OFF_S1T);   // [w][k], stride 200 halves
  _Float16* WaL  = (_Float16*)(smem + OFF_WA);    // [oc][K], stride 192
  float*    s2L  = (float*)(smem + OFF_S2);       // [oc][w]
  float*    p2L  = (float*)(smem + OFF_P2);       // [oc][32], cols 8..23 live
  float*    dL   = (float*)(smem + OFF_D);        // [oc][20]
  float*    s3L  = (float*)(smem + OFF_S3);       // [oc][20]
  float*    p3L  = (float*)(smem + OFF_P3);       // [64]
  float*    sh2L = (float*)(smem + OFF_SH2);
  float*    sc3L = (float*)(smem + OFF_SC3);
  float*    sh3L = (float*)(smem + OFF_SH3);
  float*    w3dL = (float*)(smem + OFF_W3D);
  float*    w3pL = (float*)(smem + OFF_W3P);
  float*    fcwL = (float*)(smem + OFF_FCW);
  float*    fcbL = (float*)(smem + OFF_FCB);

  const int tid  = threadIdx.x;
  const int lane = tid & 31;
  const int wv   = tid >> 5;          // wave id, 8 waves/block
  const int b    = blockIdx.x;

  // ---- one-time setup ----
  for (int i = tid; i < NK * NW; i += 256) {      // c1 tile + zero LIF1 state
    c1L[i] = c1g[b * (NK * NW) + i];
    v1L[i] = 0.f;
  }
  for (int i = tid; i < NW * 24; i += 256) {      // zero K-pad region 176..199
    int w = i / 24, k = NK + (i % 24);
    s1T[w * 200 + k] = (_Float16)0.f;
  }
  for (int i = tid; i < 16 * 32; i += 256) p2L[i] = 0.f;   // halo zero
  if (tid < 16) {
    float sc2 = g2[tid] * rsqrtf(v2g[tid] + BN_EPS);
    sh2L[tid] = b2[tid] - m2[tid] * sc2;
    float sc3 = g3[tid] * rsqrtf(v3g[tid] + BN_EPS);
    sc3L[tid] = sc3;
    sh3L[tid] = b3[tid] - m3[tid] * sc3;
  }
  if (tid < 256) { w3dL[tid] = w3d[tid]; w3pL[tid] = w3p[tid]; fcwL[tid] = fcw[tid]; }
  if (tid < 4) fcbL[tid] = fcb[tid];
  // dense conv2 weight: Wa[oc][K] = (K<176 && K/22==oc/2) ? w2[oc][K%22]*scale2 : 0
  for (int i = tid; i < 16 * NKP; i += 256) {
    int oc = i / NKP, K = i - oc * NKP;
    float val = 0.f;
    if (K < NK) {
      int ci = K / 22, h = K - ci * 22;
      if (ci == (oc >> 1))
        val = w2[oc * 22 + h] * (g2[oc] * rsqrtf(v2g[oc] + BN_EPS));
    }
    WaL[i] = (_Float16)val;
  }
  __syncthreads();

  // ---- persistent per-thread state ----
  v8f   v2acc = {};                    // LIF2 state, waves 0..3 (matches C layout)
  float v3a = 0.f, v3b = 0.f;          // LIF3 state (272 elems: tid and tid+256)
  float facc = 0.f;                    // FC accumulator, tid<4

  for (int t = 0; t < TSTEPS; ++t) {
    // -- LIF1: constant drive c1, emit f16 spikes transposed [w][k] --
    for (int i = tid; i < NK * NW; i += 256) {
      float v = v1L[i];
      v += (c1L[i] - v) * 0.5f;                    // TAU = 2
      float s = (v >= 1.0f) ? 1.0f : 0.0f;         // spike(v - V_TH)
      v1L[i] = v * (1.0f - s);
      int k = i >> 6, w = i & 63;
      s1T[w * 200 + k] = (_Float16)s;
    }
    __syncthreads();

    // -- conv2 as WMMA GEMM: out[16,64] = Wa[16,192] x S[192,64] --
    if (wv < 4) {
      const int w     = (wv << 4) + (lane & 15);   // N = w, tile per wave
      const int hi    = (lane & 16) ? 1 : 0;
      const int kbase = hi ? 16 : 0;               // B: lanes>=16 carry K=16..31
      const int M     = lane & 15;                 // A/C row
      v8f c = {};
#pragma unroll
      for (int kc = 0; kc < 6; ++kc) {
        // B fragment: 16 contiguous halves at [w][kc*32 + kbase]
        union { uvec4 u[2]; v16h h; } B;
        const uvec4* bp = (const uvec4*)(s1T + w * 200 + kc * 32 + kbase);
        B.u[0] = bp[0]; B.u[1] = bp[1];
        // A fragment matching B: halves 0..7 -> K = kbase + j,
        //                        halves 8..15 -> K = kbase + 8 + j
        const int kb8 = hi ? 8 : 0;
        v16h A;
#pragma unroll
        for (int j = 0; j < 8; ++j) {
          A[j]     = WaL[M * NKP + kc * 32 + kbase + j];
          A[j + 8] = WaL[M * NKP + kc * 32 + kbase + 8 + j];
        }
        c = __builtin_amdgcn_wmma_f32_16x16x32_f16(
                false, A, false, B.h, (short)0, c, false, false);
      }
      // -- BN shift + LIF2 in the accumulator's own lanes --
      const int mb = hi ? 8 : 0;
#pragma unroll
      for (int f = 0; f < 8; ++f) {
        int oc = mb + f;
        float cb = c[f] + sh2L[oc];
        float v  = v2acc[f];
        v += (cb - v) * 0.5f;
        float s = (v >= 1.0f) ? 1.0f : 0.0f;
        v2acc[f] = v * (1.0f - s);
        s2L[oc * 64 + w] = s;
      }
    }
    __syncthreads();

    // -- avgpool/4 -> p2 with +-8 halo for depthwise --
    {
      int oc = tid >> 4, wp = tid & 15;
      const float* r = s2L + oc * 64 + 4 * wp;
      p2L[oc * 32 + 8 + wp] = 0.25f * (r[0] + r[1] + r[2] + r[3]);
    }
    __syncthreads();

    // -- depthwise 1x16, pad 8: d[oc][wo], wo in 0..16 --
    for (int i = tid; i < 272; i += 256) {
      int oc = i / 17, wo = i - oc * 17;
      const float* p = p2L + oc * 32 + wo;
      const float* wd = w3dL + oc * 16;
      float a = 0.f;
#pragma unroll
      for (int kk = 0; kk < 16; ++kk) a = fmaf(p[kk], wd[kk], a);
      dL[oc * 20 + wo] = a;
    }
    __syncthreads();

    // -- pointwise 16x16 + BN3 + LIF3 --
    {
      int rep = 0;
      for (int i = tid; i < 272; i += 256, ++rep) {
        int oc = i / 17, wo = i - oc * 17;
        const float* wp_ = w3pL + oc * 16;
        float a = 0.f;
#pragma unroll
        for (int ic = 0; ic < 16; ++ic) a = fmaf(wp_[ic], dL[ic * 20 + wo], a);
        a = fmaf(a, sc3L[oc], sh3L[oc]);
        float v = rep ? v3b : v3a;
        v += (a - v) * 0.5f;
        float s = (v >= 1.0f) ? 1.0f : 0.0f;
        v *= (1.0f - s);
        if (rep) v3b = v; else v3a = v;
        s3L[oc * 20 + wo] = s;
      }
    }
    __syncthreads();

    // -- avgpool/4 (17 -> 4, drops tail) --
    if (tid < 64) {
      int oc = tid >> 2, wq = tid & 3;
      const float* r = s3L + oc * 20 + 4 * wq;
      p3L[tid] = 0.25f * (r[0] + r[1] + r[2] + r[3]);
    }
    __syncthreads();

    // -- FC accumulate over time --
    if (tid < 4) {
      const float* fw = fcwL + tid * 64;
      float a = 0.f;
#pragma unroll
      for (int f = 0; f < 64; ++f) a = fmaf(fw[f], p3L[f], a);
      facc += a;
    }
    __syncthreads();
  }

  if (tid < 4) out[b * 4 + tid] = facc * (1.0f / (float)TSTEPS) + fcbL[tid];
}

// ---------------- launcher ----------------
extern "C" void kernel_launch(void* const* d_in, const int* in_sizes, int n_in,
                              void* d_out, int out_size, void* d_ws, size_t ws_size,
                              hipStream_t stream) {
  const float* x    = (const float*)d_in[0];
  const float* w1   = (const float*)d_in[1];
  const float* g1   = (const float*)d_in[2];
  const float* b1   = (const float*)d_in[3];
  const float* m1   = (const float*)d_in[4];
  const float* v1   = (const float*)d_in[5];
  const float* w2   = (const float*)d_in[6];
  const float* g2   = (const float*)d_in[7];
  const float* b2   = (const float*)d_in[8];
  const float* m2   = (const float*)d_in[9];
  const float* v2   = (const float*)d_in[10];
  const float* w3d  = (const float*)d_in[11];
  const float* w3p  = (const float*)d_in[12];
  const float* g3   = (const float*)d_in[13];
  const float* b3   = (const float*)d_in[14];
  const float* m3   = (const float*)d_in[15];
  const float* v3   = (const float*)d_in[16];
  const float* fcw  = (const float*)d_in[17];
  const float* fcb  = (const float*)d_in[18];
  float* out = (float*)d_out;
  float* c1g = (float*)d_ws;                       // 128*176*64 f32 = 5.77 MB

  const int n1 = NB * NK * NW;
  eeg_conv1_bn<<<(n1 + 255) / 256, 256, 0, stream>>>(x, w1, g1, b1, m1, v1, c1g);
  eeg_snn_fused<<<NB, 256, SMEM_BYTES, stream>>>(
      c1g, w2, g2, b2, m2, v2, w3d, w3p, g3, b3, m3, v3, fcw, fcb, out);
}
